// Aligner_32289564131721
// MI455X (gfx1250) — matrix operand order repaired and checked by reading, not compiled, not run on hardware
//
#include <hip/hip_runtime.h>
#include <hip/hip_bf16.h>
#include <math.h>

// ---------------------------------------------------------------------------
// Problem constants (from reference)
// ---------------------------------------------------------------------------
#define cB 64
#define cS 512
#define cI 512
#define cH 80
#define cC 1024
#define cL 128
#define cK 31
#define cT 800

#define KX_IH  608      // (cI+cH)=592 padded to mult of 32
#define KX_HH  1024     // cC
#define KX_T1  1632     // (cI+cH+cC)=1616 padded to mult of 32
#define N3     3072     // 3*cC

typedef __attribute__((ext_vector_type(16))) _Float16 v16h;
typedef __attribute__((ext_vector_type(8)))  _Float16 v8h;
typedef __attribute__((ext_vector_type(8)))  float    v8f;

// ---------------------------------------------------------------------------
// WMMA fragment loads (CDNA5 16-bit A / B layouts, wave32)
//   A: [16 x K] f16 row-major (lda % 8 == 0)
//   W: [16 x K] f16 row-major weight tile (ldw % 32 == 0); row-major [N,K]
//      is exactly the B-fragment feed since B[k][n] = W[n][k].
// ---------------------------------------------------------------------------
__device__ __forceinline__ v16h load_afrag(const _Float16* __restrict__ aBase, int k0)
{
    v8h lo = *(const v8h*)(aBase + k0);
    v8h hi = *(const v8h*)(aBase + k0 + 8);
    return __builtin_shufflevector(lo, hi, 0,1,2,3,4,5,6,7,8,9,10,11,12,13,14,15);
}

// One wave computes a 32x32 f32 tile (2x2 WMMA tiles) of  C = A * W^T.
__device__ __forceinline__ void wmma_accum_2x2(const _Float16* __restrict__ A, int lda,
                                               const _Float16* __restrict__ W, int ldw,
                                               int kTiles,
                                               v8f& a00, v8f& a01, v8f& a10, v8f& a11)
{
    const int lane = threadIdx.x & 31;
    const int r    = lane & 15;     // A row / B column within tile
    const int hh   = lane >> 4;     // lane half
    const _Float16* aBase0 = A + r * lda + hh * 16;
    const _Float16* aBase1 = aBase0 + 16 * lda;
    const _Float16* bBase0 = W + r * ldw + hh * 16;
    const _Float16* bBase1 = bBase0 + 16 * ldw;
    for (int kt = 0; kt < kTiles; ++kt) {
        const int k0 = kt * 32;
        v16h fa0 = load_afrag(aBase0, k0);
        v16h fa1 = load_afrag(aBase1, k0);
        v16h fb0 = *(const v16h*)(bBase0 + k0);
        v16h fb1 = *(const v16h*)(bBase1 + k0);
        a00 = __builtin_amdgcn_wmma_f32_16x16x32_f16(false, fa0, false, fb0, (short)0, a00, false, false);
        a01 = __builtin_amdgcn_wmma_f32_16x16x32_f16(false, fa0, false, fb1, (short)0, a01, false, false);
        a10 = __builtin_amdgcn_wmma_f32_16x16x32_f16(false, fa1, false, fb0, (short)0, a10, false, false);
        a11 = __builtin_amdgcn_wmma_f32_16x16x32_f16(false, fa1, false, fb1, (short)0, a11, false, false);
    }
}

// Store one 16x16 tile; C pre-offset to tile origin, bias pre-offset to n0 (or null)
__device__ __forceinline__ void wmma_store(float* __restrict__ C, int ldc,
                                           const float* __restrict__ bias,
                                           const v8f& acc, bool doTanh)
{
    const int lane  = threadIdx.x & 31;
    const int n     = lane & 15;
    const int mBase = (lane >> 4) ? 8 : 0;
    float bv = bias ? bias[n] : 0.f;
#pragma unroll
    for (int j = 0; j < 8; ++j) {
        float v = acc[j] + bv;
        if (doTanh) v = tanhf(v);
        C[(mBase + j) * ldc + n] = v;
    }
}

__device__ __forceinline__ void wmma_store_2x2(float* __restrict__ C, int ldc,
                                               const float* __restrict__ bias,
                                               const v8f& a00, const v8f& a01,
                                               const v8f& a10, const v8f& a11, bool doTanh)
{
    wmma_store(C,                 ldc, bias ? bias      : nullptr, a00, doTanh);
    wmma_store(C + 16,            ldc, bias ? bias + 16 : nullptr, a01, doTanh);
    wmma_store(C + 16 * ldc,      ldc, bias ? bias      : nullptr, a10, doTanh);
    wmma_store(C + 16 * ldc + 16, ldc, bias ? bias + 16 : nullptr, a11, doTanh);
}

// ---------------------------------------------------------------------------
// Setup kernels
// ---------------------------------------------------------------------------
__global__ void k_convert_pad(const float* __restrict__ src, _Float16* __restrict__ dst,
                              int rows, int srcK, int dstK)
{
    int total = rows * dstK;
    for (int idx = blockIdx.x * blockDim.x + threadIdx.x; idx < total;
         idx += gridDim.x * blockDim.x) {
        int r = idx / dstK, k = idx - r * dstK;
        dst[idx] = (k < srcK) ? (_Float16)src[(size_t)r * srcK + k] : (_Float16)0.f;
    }
}

__global__ void k_init(float* alpha, float* align, float* hf,
                       _Float16* hh, _Float16* xcat, _Float16* xq)
{
    int idx = blockIdx.x * blockDim.x + threadIdx.x;
    if (idx < cB * cS) {
        float v = ((idx & (cS - 1)) == 0) ? 1.f : 0.f;
        alpha[idx] = v; align[idx] = v;
    }
    if (idx < cB * cC) { hf[idx] = 0.f; hh[idx] = (_Float16)0.f; }
    if (idx < cB * KX_IH) xcat[idx] = (_Float16)0.f;
    if (idx < cB * KX_T1) xq[idx] = (_Float16)0.f;
}

// key_proj[b*S+s, l] = sum_i enc[b,s,i] * wk[l,i]  (once per call)
// M=32768 (1024 m2-tiles) x N=128 (4 n2-tiles) -> 4096 waves -> 512 blocks x 8 waves
__global__ void k_keyproj(const _Float16* __restrict__ enc_h,
                          const _Float16* __restrict__ wk_h,
                          float* __restrict__ keyp)
{
    int wave = blockIdx.x * (blockDim.x >> 5) + (threadIdx.x >> 5);
    int mt2 = wave >> 2;      // 0..1023
    int nt2 = wave & 3;       // 0..3
    v8f a00 = {}, a01 = {}, a10 = {}, a11 = {};
    wmma_accum_2x2(enc_h + (size_t)mt2 * 32 * cI, cI,
                   wk_h + (size_t)nt2 * 32 * cI, cI, cI / 32, a00, a01, a10, a11);
    wmma_store_2x2(keyp + (size_t)mt2 * 32 * cL + nt2 * 32, cL, nullptr,
                   a00, a01, a10, a11, false);
}

// ---------------------------------------------------------------------------
// Per-step kernel 1: prev = alpha @ enc ; build xcat (f16) and xq (f16) prefixes
// grid: 64 blocks (b) x 256 threads
// ---------------------------------------------------------------------------
__global__ void k_prev(const float* __restrict__ alpha, const _Float16* __restrict__ enc_h,
                       const float* __restrict__ gt, int t,
                       _Float16* __restrict__ xcat, _Float16* __restrict__ xq)
{
    int b = blockIdx.x, tid = threadIdx.x;
    __shared__ float s_a[cS];
    s_a[tid]       = alpha[b * cS + tid];
    s_a[tid + 256] = alpha[b * cS + 256 + tid];
    __syncthreads();
    const _Float16* e = enc_h + (size_t)b * cS * cI;
    int i0 = tid, i1 = tid + 256;
    float acc0 = 0.f, acc1 = 0.f;
#pragma unroll 4
    for (int s = 0; s < cS; ++s) {
        float a = s_a[s];
        acc0 += a * (float)e[s * cI + i0];
        acc1 += a * (float)e[s * cI + i1];
    }
    xcat[b * KX_IH + cH + i0] = (_Float16)acc0;
    xcat[b * KX_IH + cH + i1] = (_Float16)acc1;
    xq  [b * KX_T1 + cH + i0] = (_Float16)acc0;
    xq  [b * KX_T1 + cH + i1] = (_Float16)acc1;
    if (tid < cH) {
        float f = gt[((size_t)b * cT + t) * cH + tid];
        xcat[b * KX_IH + tid] = (_Float16)f;
        xq  [b * KX_T1 + tid] = (_Float16)f;
    }
}

// ---------------------------------------------------------------------------
// Per-step kernel 2: gi = xcat @ w_ih^T + b_ih ; gh = h @ w_hh^T + b_hh
// 48 blocks x 256 threads = 384 waves; each wave one 32x32 tile.
//   gi: 2 m2-tiles x 96 n2-tiles = 192 waves, gh: 192 waves.
// ---------------------------------------------------------------------------
__global__ void k_gru_gemm(const _Float16* __restrict__ xcat, const _Float16* __restrict__ h_h,
                           const _Float16* __restrict__ wih_h, const _Float16* __restrict__ whh_h,
                           const float* __restrict__ b_ih, const float* __restrict__ b_hh,
                           float* __restrict__ gi, float* __restrict__ gh)
{
    int wave = blockIdx.x * (blockDim.x >> 5) + (threadIdx.x >> 5);
    v8f a00 = {}, a01 = {}, a10 = {}, a11 = {};
    if (wave < 192) {
        int mt2 = wave / 96, nt2 = wave % 96;
        wmma_accum_2x2(xcat + mt2 * 32 * KX_IH, KX_IH,
                       wih_h + (size_t)nt2 * 32 * KX_IH, KX_IH, KX_IH / 32,
                       a00, a01, a10, a11);
        wmma_store_2x2(gi + mt2 * 32 * N3 + nt2 * 32, N3, b_ih + nt2 * 32,
                       a00, a01, a10, a11, false);
    } else {
        wave -= 192;
        int mt2 = wave / 96, nt2 = wave % 96;
        wmma_accum_2x2(h_h + mt2 * 32 * KX_HH, KX_HH,
                       whh_h + (size_t)nt2 * 32 * KX_HH, KX_HH, KX_HH / 32,
                       a00, a01, a10, a11);
        wmma_store_2x2(gh + mt2 * 32 * N3 + nt2 * 32, N3, b_hh + nt2 * 32,
                       a00, a01, a10, a11, false);
    }
}

// ---------------------------------------------------------------------------
// Per-step kernel 3: GRU gate nonlinearity + state update
// ---------------------------------------------------------------------------
__global__ void k_gru_fin(const float* __restrict__ gi, const float* __restrict__ gh,
                          float* __restrict__ h, _Float16* __restrict__ h_h,
                          _Float16* __restrict__ xq)
{
    int idx = blockIdx.x * blockDim.x + threadIdx.x;   // 65536
    int b = idx >> 10, c = idx & (cC - 1);
    const float* gib = gi + b * N3;
    const float* ghb = gh + b * N3;
    float r = 1.f / (1.f + __expf(-(gib[c] + ghb[c])));
    float z = 1.f / (1.f + __expf(-(gib[cC + c] + ghb[cC + c])));
    float n = tanhf(gib[2 * cC + c] + r * ghb[2 * cC + c]);
    float hn = (1.f - z) * n + z * h[idx];
    h[idx] = hn;
    h_h[idx] = (_Float16)hn;
    xq[b * KX_T1 + (cI + cH) + c] = (_Float16)hn;
}

// ---------------------------------------------------------------------------
// Per-step kernel 4: t1h = tanh( xq @ t1_w^T + t1_b )  AND  qp = h @ wq^T
// t1: 2 m2 x 32 n2 = 64 waves ; qp: 2 m2 x 4 n2 = 8 waves -> 72 waves
// grid: 9 blocks x 256 threads
// ---------------------------------------------------------------------------
__global__ void k_t1_gemm(const _Float16* __restrict__ xq, const _Float16* __restrict__ t1w_h,
                          const float* __restrict__ t1_b,
                          const _Float16* __restrict__ h_h, const _Float16* __restrict__ wq_h,
                          float* __restrict__ t1h, float* __restrict__ qp)
{
    int wave = blockIdx.x * (blockDim.x >> 5) + (threadIdx.x >> 5);
    v8f a00 = {}, a01 = {}, a10 = {}, a11 = {};
    if (wave < 64) {
        int mt2 = wave / 32, nt2 = wave % 32;
        wmma_accum_2x2(xq + mt2 * 32 * KX_T1, KX_T1,
                       t1w_h + (size_t)nt2 * 32 * KX_T1, KX_T1, KX_T1 / 32,
                       a00, a01, a10, a11);
        wmma_store_2x2(t1h + mt2 * 32 * cC + nt2 * 32, cC, t1_b + nt2 * 32,
                       a00, a01, a10, a11, true);
    } else {
        wave -= 64;
        int mt2 = wave / 4, nt2 = wave % 4;
        wmma_accum_2x2(h_h + mt2 * 32 * cC, cC,
                       wq_h + (size_t)nt2 * 32 * cC, cC, cC / 32,
                       a00, a01, a10, a11);
        wmma_store_2x2(qp + mt2 * 32 * cL + nt2 * 32, cL, nullptr,
                       a00, a01, a10, a11, false);
    }
}

// ---------------------------------------------------------------------------
// Block reductions (512 threads, power of two)
// ---------------------------------------------------------------------------
__device__ __forceinline__ float blk_sum(float v, float* s_red, int tid)
{
    s_red[tid] = v; __syncthreads();
    for (int off = 256; off > 0; off >>= 1) {
        if (tid < off) s_red[tid] += s_red[tid + off];
        __syncthreads();
    }
    float r = s_red[0]; __syncthreads(); return r;
}
__device__ __forceinline__ float blk_max(float v, float* s_red, int tid)
{
    s_red[tid] = v; __syncthreads();
    for (int off = 256; off > 0; off >>= 1) {
        if (tid < off) s_red[tid] = fmaxf(s_red[tid], s_red[tid + off]);
        __syncthreads();
    }
    float r = s_red[0]; __syncthreads(); return r;
}

// ---------------------------------------------------------------------------
// Per-step kernel 5: tr = softmax(t1h@t2^T+b) ; loc-conv ; energy ;
// softmax over S ; alpha recursion. One block per batch, 512 threads.
// ---------------------------------------------------------------------------
__global__ void k_attn(const float* __restrict__ qp_g,
                       const float* __restrict__ t1h, const float* __restrict__ t2w,
                       const float* __restrict__ t2b, const float* __restrict__ keyp,
                       const float* __restrict__ convw, const float* __restrict__ convb,
                       const float* __restrict__ aggw, const float* __restrict__ aggb,
                       const float* __restrict__ mask,
                       float* __restrict__ alpha, float* __restrict__ align,
                       float* __restrict__ out, int t)
{
    int b = blockIdx.x, tid = threadIdx.x;   // tid == s
    __shared__ float s_alpha[cS];
    __shared__ float s_align[cS + 2 * (cK / 2) + 2];   // 15 halo each side
    __shared__ float s_cw[cL * cK];                    // 128*31
    __shared__ float s_qp[cL];
    __shared__ float s_aggw[cL];
    __shared__ float s_cb[cL];
    __shared__ float s_red[cS];

    // stage per-step vectors
    s_alpha[tid] = alpha[b * cS + tid];
    s_align[15 + tid] = align[b * cS + tid];
    if (tid < 15) { s_align[tid] = 0.f; s_align[15 + cS + tid] = 0.f; }
    for (int i = tid; i < cL * cK; i += 512) s_cw[i] = convw[i];
    if (tid < cL) {
        s_aggw[tid] = aggw[tid];
        s_cb[tid]   = convb[tid];
        s_qp[tid]   = qp_g[b * cL + tid];
    }

    // transition probabilities (softmax over 2)
    float p0 = 0.f, p1 = 0.f;
    {
        const float* th = t1h + b * cC;
        for (int c = tid; c < cC; c += 512) {
            float v = th[c];
            p0 += v * t2w[c];
            p1 += v * t2w[cC + c];
        }
    }
    float e0 = blk_sum(p0, s_red, tid) + t2b[0];
    float e1 = blk_sum(p1, s_red, tid) + t2b[1];
    float stop = 1.f / (1.f + __expf(e1 - e0));
    float nxt  = 1.f - stop;

    // energy for position s = tid
    int s = tid;
    const float* kp = keyp + ((size_t)b * cS + s) * cL;
    float e = aggb[0];
    for (int l = 0; l < cL; ++l) {
        float loc = s_cb[l];
        const float* cw = s_cw + l * cK;
        const float* aw = s_align + s;          // aw[k] = align[s-15+k]
#pragma unroll
        for (int k = 0; k < cK; ++k) loc += cw[k] * aw[k];
        float sc = s_qp[l] + kp[l] + loc;
        e += s_aggw[l] * tanhf(sc);
    }
    if (!(mask[b * cS + s] > 0.f)) e = -1e30f;

    // softmax over S -> new_align
    float mx = blk_max(e, s_red, tid);
    float ex = __expf(e - mx);
    float sm = blk_sum(ex, s_red, tid);
    float na_align = ex / sm;
    align[b * cS + s] = na_align;

    // alpha recursion
    float shifted = (s > 0) ? s_alpha[s - 1] : 0.f;
    float na = (stop * s_alpha[s] + nxt * shifted + 1e-5f) * na_align;
    float tot = blk_sum(na, s_red, tid);
    na /= tot;
    alpha[b * cS + s] = na;
    out[((size_t)b * cT + t) * cS + s] = na;
}

// ---------------------------------------------------------------------------
// Host launcher
// ---------------------------------------------------------------------------
extern "C" void kernel_launch(void* const* d_in, const int* in_sizes, int n_in,
                              void* d_out, int out_size, void* d_ws, size_t ws_size,
                              hipStream_t stream)
{
    const float* enc_f  = (const float*)d_in[0];
    const float* mask_f = (const float*)d_in[1];
    const float* gt_f   = (const float*)d_in[2];
    const float* wih_f  = (const float*)d_in[3];
    const float* whh_f  = (const float*)d_in[4];
    const float* bih_f  = (const float*)d_in[5];
    const float* bhh_f  = (const float*)d_in[6];
    const float* wq_f   = (const float*)d_in[7];
    const float* wk_f   = (const float*)d_in[8];
    const float* cw_f   = (const float*)d_in[9];
    const float* cb_f   = (const float*)d_in[10];
    const float* aggw_f = (const float*)d_in[11];
    const float* aggb_f = (const float*)d_in[12];
    const float* t1w_f  = (const float*)d_in[13];
    const float* t1b_f  = (const float*)d_in[14];
    const float* t2w_f  = (const float*)d_in[15];
    const float* t2b_f  = (const float*)d_in[16];
    float* out = (float*)d_out;

    // carve scratch (256B aligned)
    char* ws = (char*)d_ws;
    size_t off = 0;
    auto carve = [&](size_t bytes) -> char* {
        char* p = ws + off;
        off = (off + bytes + 255) & ~(size_t)255;
        return p;
    };
    _Float16* enc_h  = (_Float16*)carve((size_t)cB * cS * cI * 2);
    _Float16* wih_h  = (_Float16*)carve((size_t)N3 * KX_IH * 2);
    _Float16* whh_h  = (_Float16*)carve((size_t)N3 * KX_HH * 2);
    _Float16* t1w_h  = (_Float16*)carve((size_t)cC * KX_T1 * 2);
    _Float16* wk_h   = (_Float16*)carve((size_t)cL * cI * 2);
    _Float16* wq_h   = (_Float16*)carve((size_t)cL * cC * 2);
    float*    keyp   = (float*)   carve((size_t)cB * cS * cL * 4);
    _Float16* xcat_h = (_Float16*)carve((size_t)cB * KX_IH * 2);
    _Float16* xq_h   = (_Float16*)carve((size_t)cB * KX_T1 * 2);
    float*    gi     = (float*)   carve((size_t)cB * N3 * 4);
    float*    gh     = (float*)   carve((size_t)cB * N3 * 4);
    float*    h_f    = (float*)   carve((size_t)cB * cC * 4);
    _Float16* h_h    = (_Float16*)carve((size_t)cB * cC * 2);
    float*    t1h    = (float*)   carve((size_t)cB * cC * 4);
    float*    qp     = (float*)   carve((size_t)cB * cL * 4);
    float*    alpha  = (float*)   carve((size_t)cB * cS * 4);
    float*    align  = (float*)   carve((size_t)cB * cS * 4);
    (void)ws_size; (void)in_sizes; (void)n_in; (void)out_size;

    // ---- setup: f16 weight/encoding conversion, init state, key_proj ----
    k_convert_pad<<<1024, 256, 0, stream>>>(wih_f, wih_h, N3, cI + cH, KX_IH);
    k_convert_pad<<<1024, 256, 0, stream>>>(whh_f, whh_h, N3, KX_HH, KX_HH);
    k_convert_pad<<<1024, 256, 0, stream>>>(t1w_f, t1w_h, cC, cI + cH + cC, KX_T1);
    k_convert_pad<<<64,   256, 0, stream>>>(wk_f,  wk_h,  cL, cI, cI);
    k_convert_pad<<<128,  256, 0, stream>>>(wq_f,  wq_h,  cL, cC, cC);
    k_convert_pad<<<4096, 256, 0, stream>>>(enc_f, enc_h, cB * cS, cI, cI);
    k_init<<<(cB * KX_T1 + 255) / 256, 256, 0, stream>>>(alpha, align, h_f, h_h, xcat_h, xq_h);
    k_keyproj<<<512, 256, 0, stream>>>(enc_h, wk_h, keyp);

    // ---- 800 sequential attention steps ----
    for (int t = 0; t < cT; ++t) {
        k_prev<<<cB, 256, 0, stream>>>(alpha, enc_h, gt_f, t, xcat_h, xq_h);
        k_gru_gemm<<<48, 256, 0, stream>>>(xcat_h, h_h, wih_h, whh_h, bih_f, bhh_f, gi, gh);
        k_gru_fin<<<(cB * cC) / 256, 256, 0, stream>>>(gi, gh, h_f, h_h, xq_h);
        k_t1_gemm<<<9, 256, 0, stream>>>(xq_h, t1w_h, t1b_f, h_h, wq_h, t1h, qp);
        k_attn<<<cB, 512, 0, stream>>>(qp, t1h, t2w_f, t2b_f, keyp,
                                       cw_f, cb_f, aggw_f, aggb_f, mask_f,
                                       alpha, align, out, t);
    }
}
